// AttModel_9517647527942
// MI455X (gfx1250) — matrix operand order repaired and checked by reading, not compiled, not run on hardware
//
#include <hip/hip_runtime.h>
#include <hip/hip_bf16.h>
#include <math.h>

// ---------------------------------------------------------------------------
// AttModel on MI455X (gfx1250): bf16 WMMA pipeline, register-blocked.
//   h:[32,1024,256]  Wq/Wk/Wv:[512,256]  Wo:[256,512]
//   q/k/v = relu(h@W^T+b) ; att = softmax(q k^T) ; o = att v ; out = relu(o Wo^T + bo)
// ~103 GFLOP vs ~67MB true IO -> compute bound; all GEMMs on
// v_wmma_f32_16x16x32_bf16. Register blocking (1x4 output strips, Q resident
// in VGPRs, kc-outer P@V) cuts vmem issue from 4 loads/wmma to ~1.2-2.5.
// Intermediates (~145MB bf16) are L2-resident (192MB).
// ---------------------------------------------------------------------------

#define BS   32
#define SEQ  1024
#define DIN_ 256
#define HID_ 512

typedef __attribute__((ext_vector_type(16))) __bf16 v16bf;
typedef __attribute__((ext_vector_type(8)))  __bf16 v8bf;
typedef __attribute__((ext_vector_type(8)))  float  v8f;

// ---- WMMA fragment loaders -------------------------------------------------
// A (16x32 bf16, M x K), dense layout per ISA 7.12.2:
//   lanes 0-15: row M=lane, elems 0..7 = K 0..7,  elems 8..15 = K 16..23
//   lanes16-31: row M=lane-16, elems 0..7 = K 8..15, elems 8..15 = K 24..31
__device__ __forceinline__ v16bf load_a_frag(const __bf16* base, int ld, int row0, int k0) {
  const int lane = threadIdx.x & 31;
  const int row  = row0 + (lane & 15);
  const int hi   = lane >> 4;
  const __bf16* p = base + (size_t)row * ld + k0 + hi * 8;
  v8bf lo  = *(const v8bf*)(p);
  v8bf hi8 = *(const v8bf*)(p + 16);
  v16bf r;
#pragma unroll
  for (int i = 0; i < 8; ++i) { r[i] = lo[i]; r[8 + i] = hi8[i]; }
  return r;
}

// B (32x16 bf16, K x N): lane = column, same K order as A fragment
// (lanes 0-15: K 0..7,16..23; lanes 16-31: K 8..15,24..31). `base` holds B^T row-major.
__device__ __forceinline__ v16bf load_b_frag(const __bf16* base, int ld, int col0, int k0) {
  const int lane = threadIdx.x & 31;
  const int col  = col0 + (lane & 15);
  const int kk   = k0 + 8 * (lane >> 4);
  const __bf16* p = base + (size_t)col * ld + kk;
  v8bf lo  = *(const v8bf*)(p);
  v8bf hi8 = *(const v8bf*)(p + 16);
  v16bf r;
#pragma unroll
  for (int i = 0; i < 8; ++i) { r[i] = lo[i]; r[8 + i] = hi8[i]; }
  return r;
}

__device__ __forceinline__ v8f wmma_bf16(v16bf a, v16bf b, v8f c) {
  return __builtin_amdgcn_wmma_f32_16x16x32_bf16(false, a, false, b, (short)0, c, false, false);
}

// ---- Kernel 1: fp32 -> bf16 pack ------------------------------------------
__global__ void __launch_bounds__(256) pack_bf16_kernel(
    const float* __restrict__ h,
    const float* __restrict__ Wv, const float* __restrict__ Wk,
    const float* __restrict__ Wq, const float* __restrict__ Wo,
    __bf16* __restrict__ hb,
    __bf16* __restrict__ wvb, __bf16* __restrict__ wkb,
    __bf16* __restrict__ wqb, __bf16* __restrict__ wob) {
  const size_t stride = (size_t)gridDim.x * blockDim.x;
  size_t i = (size_t)blockIdx.x * blockDim.x + threadIdx.x;
  const size_t NH = (size_t)BS * SEQ * DIN_;
  for (size_t x = i; x < NH; x += stride) hb[x] = (__bf16)h[x];
  const size_t WSZ = (size_t)HID_ * DIN_;   // all four weight mats are 512*256
  for (size_t x = i; x < WSZ; x += stride) {
    wvb[x] = (__bf16)Wv[x];
    wkb[x] = (__bf16)Wk[x];
    wqb[x] = (__bf16)Wq[x];
    wob[x] = (__bf16)Wo[x];
  }
}

// ---- Kernel 2: fused Q/K/V projections ------------------------------------
// One wave per 16x64 output strip (4 tiles): A fragment loaded once per
// K-chunk, reused by 4 WMMAs. q,k -> [B][N][HID]; v -> transposed [B][HID][N].
__global__ void __launch_bounds__(128) proj_qkv_kernel(
    const __bf16* __restrict__ hb,
    const __bf16* __restrict__ wqb, const __bf16* __restrict__ wkb,
    const __bf16* __restrict__ wvb,
    const float* __restrict__ bq, const float* __restrict__ bk,
    const float* __restrict__ bv,
    __bf16* __restrict__ qb, __bf16* __restrict__ kb, __bf16* __restrict__ vtb) {
  const int wave  = blockIdx.x * (blockDim.x >> 5) + (threadIdx.x >> 5);
  const int STRIPS = BS * (SEQ / 16) * (HID_ / 64);       // 16384 per projection
  const int which = wave / STRIPS;                        // 0=q 1=k 2=v
  int t = wave % STRIPS;
  const int b  = t / ((SEQ / 16) * (HID_ / 64));
  t            = t % ((SEQ / 16) * (HID_ / 64));
  const int n0 = (t / (HID_ / 64)) * 16;
  const int h0 = (t % (HID_ / 64)) * 64;

  const __bf16* W    = (which == 0) ? wqb : (which == 1) ? wkb : wvb;
  const float*  bias = (which == 0) ? bq  : (which == 1) ? bk  : bv;
  const __bf16* A    = hb + (size_t)b * SEQ * DIN_;

  v8f c[4];
#pragma unroll
  for (int u = 0; u < 4; ++u) c[u] = (v8f){0.f,0.f,0.f,0.f,0.f,0.f,0.f,0.f};
#pragma unroll
  for (int k0 = 0; k0 < DIN_; k0 += 32) {
    v16bf a = load_a_frag(A, DIN_, n0, k0);        // shared by 4 tiles
#pragma unroll
    for (int u = 0; u < 4; ++u) {
      v16bf bb = load_b_frag(W, DIN_, h0 + u * 16, k0);
      c[u] = wmma_bf16(a, bb, c[u]);
    }
  }

  const int lane = threadIdx.x & 31;
  const int col  = lane & 15;           // N (hid) column within a tile
  const int rb   = (lane >> 4) * 8;     // row base (M) for this half-wave
  if (which < 2) {
    __bf16* O = ((which == 0) ? qb : kb) + (size_t)b * SEQ * HID_;
#pragma unroll
    for (int u = 0; u < 4; ++u) {
      const float bi = bias[h0 + u * 16 + col];
#pragma unroll
      for (int j = 0; j < 8; ++j) {
        float x = fmaxf(c[u][j] + bi, 0.f);
        O[(size_t)(n0 + rb + j) * HID_ + h0 + u * 16 + col] = (__bf16)x;
      }
    }
  } else {
    // transposed store: rows = hid col, cols = sequence (contiguous over j)
#pragma unroll
    for (int u = 0; u < 4; ++u) {
      const float bi = bias[h0 + u * 16 + col];
      v8bf st;
#pragma unroll
      for (int j = 0; j < 8; ++j) st[j] = (__bf16)fmaxf(c[u][j] + bi, 0.f);
      *(v8bf*)(vtb + (size_t)b * HID_ * SEQ +
               (size_t)(h0 + u * 16 + col) * SEQ + n0 + rb) = st;
    }
  }
}

// ---- Kernel 3: flash attention (online softmax) ----------------------------
// 128 threads (4 waves) per (batch, 16-query-row tile). Q fragments stay
// resident in VGPRs (16 x v16bf). Per 128-wide key chunk: wave w computes 2
// S tiles (keys [w*32, w*32+32)) reusing each Q fragment; waves cooperate on
// softmax stats via LDS; P staged bf16 in LDS; each wave then does P@V for
// its private 128-column HID slice with the P fragment hoisted kc-outer.
__global__ void __launch_bounds__(128) flash_attn_kernel(
    const __bf16* __restrict__ qb, const __bf16* __restrict__ kb,
    const __bf16* __restrict__ vtb, __bf16* __restrict__ ob) {
  __shared__ __bf16 Pl[16 * 128];          // P tile [16 rows][128 keys]
  __shared__ float wmaxs[4][16];
  __shared__ float wsums[4][16];
  __shared__ float rmax[16], rsum[16], rscale[16];

  const int b  = blockIdx.x / (SEQ / 16);
  const int n0 = (blockIdx.x % (SEQ / 16)) * 16;
  const int w    = threadIdx.x >> 5;
  const int lane = threadIdx.x & 31;
  const int col  = lane & 15;
  const int rb   = (lane >> 4) * 8;

  const __bf16* Qb  = qb  + (size_t)b * SEQ * HID_;
  const __bf16* Kb  = kb  + (size_t)b * SEQ * HID_;
  const __bf16* Vtb = vtb + (size_t)b * HID_ * SEQ;

  if (threadIdx.x < 16) { rmax[threadIdx.x] = -INFINITY; rsum[threadIdx.x] = 0.f; }
  __syncthreads();

  // Q tile resident in registers: 16 K-chunks of 32 -> 128 VGPRs
  v16bf qf[HID_ / 32];
#pragma unroll
  for (int kc = 0; kc < HID_ / 32; ++kc)
    qf[kc] = load_a_frag(Qb, HID_, n0, kc * 32);

  v8f acc[8];   // O accumulator: 16 rows x 128 cols (this wave's HID slice)
#pragma unroll
  for (int t = 0; t < 8; ++t) acc[t] = (v8f){0.f,0.f,0.f,0.f,0.f,0.f,0.f,0.f};

  for (int mb = 0; mb < SEQ / 128; ++mb) {
    const int mbase = mb * 128;

    // ---- 2 S tiles per wave: key cols mbase + w*32 + {0,16} --------------
    v8f s[2];
    s[0] = (v8f){0.f,0.f,0.f,0.f,0.f,0.f,0.f,0.f};
    s[1] = s[0];
#pragma unroll
    for (int kc = 0; kc < HID_ / 32; ++kc) {
#pragma unroll
      for (int u = 0; u < 2; ++u) {
        v16bf bf = load_b_frag(Kb, HID_, mbase + w * 32 + u * 16, kc * 32);
        s[u] = wmma_bf16(qf[kc], bf, s[u]);
      }
    }

    // ---- per-row max over this wave's 32 columns -------------------------
    float mloc[8];
#pragma unroll
    for (int j = 0; j < 8; ++j) mloc[j] = fmaxf(s[0][j], s[1][j]);
#pragma unroll
    for (int m = 1; m < 16; m <<= 1)
#pragma unroll
      for (int j = 0; j < 8; ++j) mloc[j] = fmaxf(mloc[j], __shfl_xor(mloc[j], m, 32));
    if (col == 0)
#pragma unroll
      for (int j = 0; j < 8; ++j) wmaxs[w][rb + j] = mloc[j];
    __syncthreads();

    // ---- combine 4 waves, update running max, compute O rescale ----------
    if (threadIdx.x < 16) {
      const int r = threadIdx.x;
      float cm = fmaxf(fmaxf(wmaxs[0][r], wmaxs[1][r]), fmaxf(wmaxs[2][r], wmaxs[3][r]));
      float nm = fmaxf(rmax[r], cm);
      rscale[r] = __expf(rmax[r] - nm);     // 0 on first chunk (-inf - nm)
      rmax[r]   = nm;
      rsum[r]  *= rscale[r];
    }
    __syncthreads();

    // ---- P = exp(S - max): stage bf16 tile in LDS, reduce row sums -------
    float mx[8], sl[8];
#pragma unroll
    for (int j = 0; j < 8; ++j) { mx[j] = rmax[rb + j]; sl[j] = 0.f; }
#pragma unroll
    for (int u = 0; u < 2; ++u)
#pragma unroll
      for (int j = 0; j < 8; ++j) {
        float p = __expf(s[u][j] - mx[j]);
        sl[j] += p;
        Pl[(rb + j) * 128 + w * 32 + u * 16 + col] = (__bf16)p;
      }
#pragma unroll
    for (int m = 1; m < 16; m <<= 1)
#pragma unroll
      for (int j = 0; j < 8; ++j) sl[j] += __shfl_xor(sl[j], m, 32);
    if (col == 0)
#pragma unroll
      for (int j = 0; j < 8; ++j) wsums[w][rb + j] = sl[j];
    // rescale O accumulators by exp(old_max - new_max)
    float sc[8];
#pragma unroll
    for (int j = 0; j < 8; ++j) sc[j] = rscale[rb + j];
#pragma unroll
    for (int t = 0; t < 8; ++t)
#pragma unroll
      for (int j = 0; j < 8; ++j) acc[t][j] *= sc[j];
    __syncthreads();

    if (threadIdx.x < 16) {
      const int r = threadIdx.x;
      rsum[r] += wsums[0][r] + wsums[1][r] + wsums[2][r] + wsums[3][r];
    }

    // ---- O += P @ V, P fragment hoisted (kc outer) -----------------------
#pragma unroll
    for (int kc = 0; kc < 4; ++kc) {
      v16bf a = load_a_frag((const __bf16*)Pl, 128, 0, kc * 32);  // ds loads
#pragma unroll
      for (int t = 0; t < 8; ++t) {
        v16bf bf = load_b_frag(Vtb, SEQ, w * 128 + t * 16, mbase + kc * 32);
        acc[t] = wmma_bf16(a, bf, acc[t]);
      }
    }
    __syncthreads();
  }

  // ---- finalize: O /= rowsum, store bf16 [B][N][HID] ----------------------
  float inv[8];
#pragma unroll
  for (int j = 0; j < 8; ++j) inv[j] = 1.0f / rsum[rb + j];
  __bf16* Ob = ob + (size_t)b * SEQ * HID_;
#pragma unroll
  for (int t = 0; t < 8; ++t) {
    const int h0 = w * 128 + t * 16;
#pragma unroll
    for (int j = 0; j < 8; ++j)
      Ob[(size_t)(n0 + rb + j) * HID_ + h0 + col] = (__bf16)(acc[t][j] * inv[j]);
  }
}

// ---- Kernel 4: output projection, relu, fp32 store -------------------------
// One wave per 16x64 output strip (4 tiles), A fragment reused 4x.
__global__ void __launch_bounds__(128) out_proj_kernel(
    const __bf16* __restrict__ ob, const __bf16* __restrict__ wob,
    const float* __restrict__ bo, float* __restrict__ out) {
  const int wave  = blockIdx.x * (blockDim.x >> 5) + (threadIdx.x >> 5);
  int t = wave;
  const int b  = t / ((SEQ / 16) * (DIN_ / 64));
  t            = t % ((SEQ / 16) * (DIN_ / 64));
  const int n0 = (t / (DIN_ / 64)) * 16;
  const int d0 = (t % (DIN_ / 64)) * 64;

  const __bf16* A = ob + (size_t)b * SEQ * HID_;
  v8f c[4];
#pragma unroll
  for (int u = 0; u < 4; ++u) c[u] = (v8f){0.f,0.f,0.f,0.f,0.f,0.f,0.f,0.f};
#pragma unroll
  for (int k0 = 0; k0 < HID_; k0 += 32) {
    v16bf a = load_a_frag(A, HID_, n0, k0);
#pragma unroll
    for (int u = 0; u < 4; ++u) {
      v16bf bb = load_b_frag(wob, HID_, d0 + u * 16, k0);  // Wo [DIN][HID] == B^T
      c[u] = wmma_bf16(a, bb, c[u]);
    }
  }
  const int lane = threadIdx.x & 31;
  const int col  = lane & 15;
  const int rb   = (lane >> 4) * 8;
  float* O = out + (size_t)b * SEQ * DIN_;
#pragma unroll
  for (int u = 0; u < 4; ++u) {
    const float bi = bo[d0 + u * 16 + col];
#pragma unroll
    for (int j = 0; j < 8; ++j)
      O[(size_t)(n0 + rb + j) * DIN_ + d0 + u * 16 + col] = fmaxf(c[u][j] + bi, 0.f);
  }
}

// ---------------------------------------------------------------------------
extern "C" void kernel_launch(void* const* d_in, const int* in_sizes, int n_in,
                              void* d_out, int out_size, void* d_ws, size_t ws_size,
                              hipStream_t stream) {
  const float* h  = (const float*)d_in[0];
  const float* Wv = (const float*)d_in[1];
  const float* bv = (const float*)d_in[2];
  const float* Wk = (const float*)d_in[3];
  const float* bk = (const float*)d_in[4];
  const float* Wq = (const float*)d_in[5];
  const float* bq = (const float*)d_in[6];
  const float* Wo = (const float*)d_in[7];
  const float* bo = (const float*)d_in[8];
  float* out = (float*)d_out;

  // workspace layout (bf16 buffers; ~145 MB total, L2-resident)
  char* ws = (char*)d_ws;
  size_t off = 0;
  auto take = [&](size_t bytes) { char* p = ws + off; off += bytes; return p; };
  __bf16* hb  = (__bf16*)take((size_t)BS * SEQ * DIN_ * 2);
  __bf16* wqb = (__bf16*)take((size_t)HID_ * DIN_ * 2);
  __bf16* wkb = (__bf16*)take((size_t)HID_ * DIN_ * 2);
  __bf16* wvb = (__bf16*)take((size_t)HID_ * DIN_ * 2);
  __bf16* wob = (__bf16*)take((size_t)DIN_ * HID_ * 2);
  __bf16* qb  = (__bf16*)take((size_t)BS * SEQ * HID_ * 2);
  __bf16* kb  = (__bf16*)take((size_t)BS * SEQ * HID_ * 2);
  __bf16* vtb = (__bf16*)take((size_t)BS * HID_ * SEQ * 2);  // transposed V
  __bf16* ob  = (__bf16*)take((size_t)BS * SEQ * HID_ * 2);
  (void)ws_size; (void)in_sizes; (void)n_in; (void)out_size;

  pack_bf16_kernel<<<2048, 256, 0, stream>>>(h, Wv, Wk, Wq, Wo,
                                             hb, wvb, wkb, wqb, wob);

  // 3 projections * 16384 strips, 4 waves/block
  proj_qkv_kernel<<<(3 * BS * (SEQ / 16) * (HID_ / 64)) / 4, 128, 0, stream>>>(
      hb, wqb, wkb, wvb, bq, bk, bv, qb, kb, vtb);

  flash_attn_kernel<<<BS * (SEQ / 16), 128, 0, stream>>>(qb, kb, vtb, ob);

  out_proj_kernel<<<(BS * (SEQ / 16) * (DIN_ / 64)) / 4, 128, 0, stream>>>(
      ob, wob, bo, out);
}
